// GAT_18992345383339
// MI455X (gfx1250) — matrix operand order repaired and checked by reading, not compiled, not run on hardware
//
#include <hip/hip_runtime.h>
#include <math.h>

#define N_NODES 50000
#define N_EDGES 800000
#define ETOT (N_EDGES + N_NODES)   // self loops appended
#define D_IN 256
#define HID 64
#define HEADS 4
#define D_H1 256                   // HEADS*HID
#define OUT_C 40
#define OUT_P 48                   // padded to 3 WMMA col tiles
#define NEG_SLOPE 0.2f
#define KSTEPS 8                   // 256 / 32

typedef __attribute__((ext_vector_type(16))) _Float16 v16h;
typedef __attribute__((ext_vector_type(8)))  float    v8f;

// ---------- helpers ----------
// order-preserving float -> uint map so u32 atomicMax implements float max
__device__ __forceinline__ unsigned fmap(float f) {
  unsigned u = __float_as_uint(f);
  return (u & 0x80000000u) ? ~u : (u | 0x80000000u);
}
__device__ __forceinline__ float funmap(unsigned u) {
  return (u & 0x80000000u) ? __uint_as_float(u & 0x7FFFFFFFu)
                           : __uint_as_float(~u);
}
__device__ __forceinline__ void edge_sd(const int* ei, int e, int& s, int& d) {
  if (e < N_EDGES) { s = ei[e]; d = ei[N_EDGES + e]; }
  else             { s = d = e - N_EDGES; }          // self loop
}
__device__ __forceinline__ float lrelu(float x) {
  return x > 0.f ? x : NEG_SLOPE * x;
}
// WMMA 16-bit A/B fragment coordinates: for K-offset ko (0..31) within a step,
// lane-halves are: hi = (ko>>3)&1 ; j = (ko&7) + (ko>=16 ? 8 : 0) ; lane = idx + 16*hi
__device__ __forceinline__ void frag_pos(int ko, int& hi, int& j) {
  hi = (ko >> 3) & 1;
  j  = (ko & 7) + ((ko >> 4) << 3);
}

// ---------- weight conversion: pack into exact B-fragment order ----------
// packed index p: j = p&15, lane = (p>>4)&31, s = (p>>9)&7, t = p>>12
// source: k = s*32 + (lane>>4)*8 + (j&7) + (j>>3)*16 ; col = t*16 + (lane&15)
__global__ void prep_weights_k(const float* __restrict__ W1,
                               const float* __restrict__ W2,
                               _Float16* __restrict__ w1p,
                               _Float16* __restrict__ w2p) {
  int i = blockIdx.x * blockDim.x + threadIdx.x;
  int stride = gridDim.x * blockDim.x;
  for (int p = i; p < D_IN * D_H1; p += stride) {          // 16 tiles * 8 * 32 * 16
    int j = p & 15, lane = (p >> 4) & 31, s = (p >> 9) & 7, t = p >> 12;
    int k = s * 32 + (lane >> 4) * 8 + (j & 7) + (j >> 3) * 16;
    int col = t * 16 + (lane & 15);
    w1p[p] = (_Float16)W1[k * D_H1 + col];
  }
  for (int p = i; p < KSTEPS * 32 * 16 * 3; p += stride) { // 3 tiles * 8 * 32 * 16
    int j = p & 15, lane = (p >> 4) & 31, s = (p >> 9) & 7, t = p >> 12;
    int k = s * 32 + (lane >> 4) * 8 + (j & 7) + (j >> 3) * 16;
    int col = t * 16 + (lane & 15);
    w2p[p] = (col < OUT_C) ? (_Float16)W2[k * OUT_C + col] : (_Float16)0.f;
  }
}

// ---------- GEMM1: h1[N,256] = x[N,256] @ W1, f16 WMMA / f32 acc ----------
// LDS holds the 16-row x strip pre-swizzled into A-fragment order:
// Ap[(s*32 + lane)*16 + j]  ->  one 32B ds load per K-step per lane.
__global__ void gemm1_k(const float* __restrict__ X,
                        const _Float16* __restrict__ Wp,
                        float* __restrict__ H) {
  __shared__ __attribute__((aligned(32))) _Float16 Ap[KSTEPS * 32 * 16]; // 8KB
  const int tid  = threadIdx.x;
  const int row0 = blockIdx.x * 16;
  for (int i = tid; i < 16 * D_IN; i += 512) {
    int r = i >> 8, c = i & 255;
    int s = c >> 5, ko = c & 31, hi, j;
    frag_pos(ko, hi, j);
    Ap[((s * 32 + (r + (hi << 4))) << 4) + j] =
        (_Float16)X[(size_t)(row0 + r) * D_IN + c];
  }
  __syncthreads();

  const int lane = tid & 31;
  const int m    = lane & 15;
  const int hi   = lane >> 4;
  const int ct   = tid >> 5;              // 16 waves -> 16 col tiles
  const _Float16* bp = Wp + ((size_t)(ct * KSTEPS) << 9); // *32*16
  v8f acc = {};
  #pragma unroll
  for (int s = 0; s < KSTEPS; ++s) {
    v16h a = *(const v16h*)(Ap + ((s * 32 + lane) << 4));
    v16h b = *(const v16h*)(bp + (((s * 32) + lane) << 4));
    acc = __builtin_amdgcn_wmma_f32_16x16x32_f16(false, a, false, b,
                                                 (short)0, acc, false, false);
  }
  #pragma unroll
  for (int v = 0; v < 8; ++v)             // D: row = v + hi*8, col = m
    H[(size_t)(row0 + hi * 8 + v) * D_H1 + ct * 16 + m] = acc[v];
}

// ---------- GEMM2: h2[N,48] = elu_out[N,256] @ W2pad ----------
__global__ void gemm2_k(const float* __restrict__ X,
                        const _Float16* __restrict__ Wp,
                        float* __restrict__ H) {
  __shared__ __attribute__((aligned(32))) _Float16 Ap[KSTEPS * 32 * 16];
  const int tid  = threadIdx.x;
  const int row0 = blockIdx.x * 16;
  for (int i = tid; i < 16 * D_H1; i += 128) {
    int r = i >> 8, c = i & 255;
    int s = c >> 5, ko = c & 31, hi, j;
    frag_pos(ko, hi, j);
    Ap[((s * 32 + (r + (hi << 4))) << 4) + j] =
        (_Float16)X[(size_t)(row0 + r) * D_H1 + c];
  }
  __syncthreads();

  const int ct = tid >> 5;
  if (ct < 3) {                           // uniform per wave: EXEC stays all-ones
    const int lane = tid & 31;
    const int m = lane & 15, hi = lane >> 4;
    const _Float16* bp = Wp + ((size_t)(ct * KSTEPS) << 9);
    v8f acc = {};
    #pragma unroll
    for (int s = 0; s < KSTEPS; ++s) {
      v16h a = *(const v16h*)(Ap + ((s * 32 + lane) << 4));
      v16h b = *(const v16h*)(bp + (((s * 32) + lane) << 4));
      acc = __builtin_amdgcn_wmma_f32_16x16x32_f16(false, a, false, b,
                                                   (short)0, acc, false, false);
    }
    #pragma unroll
    for (int v = 0; v < 8; ++v)
      H[(size_t)(row0 + hi * 8 + v) * OUT_P + ct * 16 + m] = acc[v];
  }
}

// ---------- layer-1 attention coefficients: warp per (node, head) ----------
__global__ void attn1_k(const float* __restrict__ H,
                        const float* __restrict__ att_src,
                        const float* __restrict__ att_dst,
                        float* __restrict__ asrc, float* __restrict__ adst) {
  int g = blockIdx.x * 8 + (threadIdx.x >> 5);
  if (g >= N_NODES * HEADS) return;
  int n = g >> 2, h = g & 3;
  int lane = threadIdx.x & 31;
  const float* hp = H + (size_t)n * D_H1 + h * HID;
  float v0 = hp[lane], v1 = hp[lane + 32];
  float s = v0 * att_src[h * HID + lane] + v1 * att_src[h * HID + lane + 32];
  float d = v0 * att_dst[h * HID + lane] + v1 * att_dst[h * HID + lane + 32];
  #pragma unroll
  for (int off = 16; off; off >>= 1) {
    s += __shfl_down(s, off, 32);
    d += __shfl_down(d, off, 32);
  }
  if (lane == 0) { asrc[g] = s; adst[g] = d; }
}

// ---------- layer-1 edge passes ----------
__global__ void ep1_l1(const int* __restrict__ ei, const float* __restrict__ as,
                       const float* __restrict__ ad, unsigned* __restrict__ mu) {
  int e = blockIdx.x * blockDim.x + threadIdx.x;
  if (e >= ETOT) return;
  int s, d; edge_sd(ei, e, s, d);
  #pragma unroll
  for (int h = 0; h < HEADS; ++h) {
    float l = lrelu(as[s * HEADS + h] + ad[d * HEADS + h]);
    atomicMax(&mu[d * HEADS + h], fmap(l));
  }
}
__global__ void ep2_l1(const int* __restrict__ ei, const float* __restrict__ as,
                       const float* __restrict__ ad, const unsigned* __restrict__ mu,
                       float* __restrict__ denom) {
  int e = blockIdx.x * blockDim.x + threadIdx.x;
  if (e >= ETOT) return;
  int s, d; edge_sd(ei, e, s, d);
  #pragma unroll
  for (int h = 0; h < HEADS; ++h) {
    float l = lrelu(as[s * HEADS + h] + ad[d * HEADS + h]);
    atomicAdd(&denom[d * HEADS + h], expf(l - funmap(mu[d * HEADS + h])));
  }
}
__global__ void ep3_l1(const int* __restrict__ ei, const float* __restrict__ as,
                       const float* __restrict__ ad, const unsigned* __restrict__ mu,
                       const float* __restrict__ denom, const float* __restrict__ H,
                       float* __restrict__ out) {
  int w = blockIdx.x * 8 + (threadIdx.x >> 5);   // warp per edge
  if (w >= ETOT) return;
  int lane = threadIdx.x & 31;
  int s, d; edge_sd(ei, w, s, d);
  float alpha[HEADS];
  #pragma unroll
  for (int h = 0; h < HEADS; ++h) {
    float l = lrelu(as[s * HEADS + h] + ad[d * HEADS + h]);
    alpha[h] = expf(l - funmap(mu[d * HEADS + h])) / denom[d * HEADS + h];
  }
  const float* hs = H + (size_t)s * D_H1;
  float* op = out + (size_t)d * D_H1;
  #pragma unroll
  for (int i = 0; i < 8; ++i) {                  // coalesced gather + atomic scatter
    int c = lane + 32 * i;
    atomicAdd(&op[c], hs[c] * alpha[c >> 6]);
  }
}

// ---------- ELU (bias b1 applied first, matching reference) ----------
__global__ void elu_k(float* __restrict__ o, const float* __restrict__ b1) {
  size_t i = (size_t)blockIdx.x * blockDim.x + threadIdx.x;
  if (i >= (size_t)N_NODES * D_H1) return;
  float v = o[i] + b1[i & 255];
  o[i] = v > 0.f ? v : (expf(v) - 1.f);
}

// ---------- layer-2 attention coefficients: warp per node ----------
__global__ void attn2_k(const float* __restrict__ H, const float* __restrict__ att_src,
                        const float* __restrict__ att_dst,
                        float* __restrict__ asrc, float* __restrict__ adst) {
  int n = blockIdx.x * 8 + (threadIdx.x >> 5);
  if (n >= N_NODES) return;
  int lane = threadIdx.x & 31;
  const float* hp = H + (size_t)n * OUT_P;
  float s = hp[lane] * att_src[lane];
  float d = hp[lane] * att_dst[lane];
  if (lane < OUT_C - 32) {
    s += hp[lane + 32] * att_src[lane + 32];
    d += hp[lane + 32] * att_dst[lane + 32];
  }
  #pragma unroll
  for (int off = 16; off; off >>= 1) {
    s += __shfl_down(s, off, 32);
    d += __shfl_down(d, off, 32);
  }
  if (lane == 0) { asrc[n] = s; adst[n] = d; }
}

// ---------- layer-2 edge passes (single head) ----------
__global__ void ep1_l2(const int* __restrict__ ei, const float* __restrict__ as,
                       const float* __restrict__ ad, unsigned* __restrict__ mu) {
  int e = blockIdx.x * blockDim.x + threadIdx.x;
  if (e >= ETOT) return;
  int s, d; edge_sd(ei, e, s, d);
  atomicMax(&mu[d], fmap(lrelu(as[s] + ad[d])));
}
__global__ void ep2_l2(const int* __restrict__ ei, const float* __restrict__ as,
                       const float* __restrict__ ad, const unsigned* __restrict__ mu,
                       float* __restrict__ denom) {
  int e = blockIdx.x * blockDim.x + threadIdx.x;
  if (e >= ETOT) return;
  int s, d; edge_sd(ei, e, s, d);
  atomicAdd(&denom[d], expf(lrelu(as[s] + ad[d]) - funmap(mu[d])));
}
__global__ void ep3_l2(const int* __restrict__ ei, const float* __restrict__ as,
                       const float* __restrict__ ad, const unsigned* __restrict__ mu,
                       const float* __restrict__ denom, const float* __restrict__ H,
                       float* __restrict__ out) {
  int w = blockIdx.x * 8 + (threadIdx.x >> 5);
  if (w >= ETOT) return;
  int lane = threadIdx.x & 31;
  int s, d; edge_sd(ei, w, s, d);
  float alpha = expf(lrelu(as[s] + ad[d]) - funmap(mu[d])) / denom[d];
  const float* hs = H + (size_t)s * OUT_P;
  float* op = out + (size_t)d * OUT_C;
  atomicAdd(&op[lane], hs[lane] * alpha);
  if (lane < OUT_C - 32) atomicAdd(&op[lane + 32], hs[lane + 32] * alpha);
}

// ---------- final: bias + log_softmax over 40 classes (warp per node) ----------
__global__ void final_k(const float* __restrict__ out2, const float* __restrict__ b2,
                        float* __restrict__ y) {
  int n = blockIdx.x * 8 + (threadIdx.x >> 5);
  if (n >= N_NODES) return;
  int lane = threadIdx.x & 31;
  const float* op = out2 + (size_t)n * OUT_C;
  float v0 = op[lane] + b2[lane];
  float v1 = (lane < OUT_C - 32) ? op[lane + 32] + b2[lane + 32] : -1e30f;
  float mx = fmaxf(v0, v1);
  #pragma unroll
  for (int off = 16; off; off >>= 1) mx = fmaxf(mx, __shfl_xor(mx, off, 32));
  float se = expf(v0 - mx) + ((lane < OUT_C - 32) ? expf(v1 - mx) : 0.f);
  #pragma unroll
  for (int off = 16; off; off >>= 1) se += __shfl_xor(se, off, 32);
  float lse = mx + logf(se);
  y[(size_t)n * OUT_C + lane] = v0 - lse;
  if (lane < OUT_C - 32) y[(size_t)n * OUT_C + lane + 32] = v1 - lse;
}

extern "C" void kernel_launch(void* const* d_in, const int* in_sizes, int n_in,
                              void* d_out, int out_size, void* d_ws, size_t ws_size,
                              hipStream_t stream) {
  const float* x        = (const float*)d_in[0];
  const int*   ei       = (const int*)d_in[1];       // [2,E]
  const float* W1       = (const float*)d_in[2];
  const float* att_s1   = (const float*)d_in[3];
  const float* att_d1   = (const float*)d_in[4];
  const float* b1       = (const float*)d_in[5];
  const float* W2       = (const float*)d_in[6];
  const float* att_s2   = (const float*)d_in[7];
  const float* att_d2   = (const float*)d_in[8];
  const float* b2       = (const float*)d_in[9];
  float* y = (float*)d_out;

  // ---- workspace layout (float units; w1p/w2p offsets are 32B aligned) ----
  float* f = (float*)d_ws;
  float*    h1     = f;                              // N*256
  float*    out1   = h1    + (size_t)N_NODES * D_H1; // N*256
  float*    as1    = out1  + (size_t)N_NODES * D_H1; // N*4
  float*    ad1    = as1   + (size_t)N_NODES * HEADS;
  unsigned* m1u    = (unsigned*)(ad1 + (size_t)N_NODES * HEADS);
  float*    den1   = (float*)(m1u + (size_t)N_NODES * HEADS);
  _Float16* w1p    = (_Float16*)(den1 + (size_t)N_NODES * HEADS);        // 256*256 h
  _Float16* w2p    = w1p + (size_t)D_IN * D_H1;                          // 256*48 h
  // layer-2 buffers alias the (dead after ep3_l1/elu input consumed) h1 region
  float*    h2     = h1;                             // N*48
  float*    as2    = h2  + (size_t)N_NODES * OUT_P;  // N
  float*    ad2    = as2 + N_NODES;
  unsigned* m2u    = (unsigned*)(ad2 + N_NODES);
  float*    den2   = (float*)(m2u + N_NODES);
  float*    out2   = den2 + N_NODES;                 // N*40

  // ---- layer 1 ----
  prep_weights_k<<<256, 256, 0, stream>>>(W1, W2, w1p, w2p);
  hipMemsetAsync(out1, 0, (size_t)N_NODES * D_H1 * sizeof(float), stream);
  hipMemsetAsync(m1u,  0, (size_t)N_NODES * HEADS * sizeof(unsigned), stream);
  hipMemsetAsync(den1, 0, (size_t)N_NODES * HEADS * sizeof(float), stream);

  gemm1_k<<<N_NODES / 16, 512, 0, stream>>>(x, w1p, h1);
  attn1_k<<<(N_NODES * HEADS) / 8, 256, 0, stream>>>(h1, att_s1, att_d1, as1, ad1);
  ep1_l1<<<(ETOT + 255) / 256, 256, 0, stream>>>(ei, as1, ad1, m1u);
  ep2_l1<<<(ETOT + 255) / 256, 256, 0, stream>>>(ei, as1, ad1, m1u, den1);
  ep3_l1<<<ETOT / 8, 256, 0, stream>>>(ei, as1, ad1, m1u, den1, h1, out1);
  elu_k<<<(N_NODES * D_H1) / 256, 256, 0, stream>>>(out1, b1);

  // ---- layer 2 (h1 region now reusable) ----
  hipMemsetAsync(m2u,  0, (size_t)N_NODES * sizeof(unsigned), stream);
  hipMemsetAsync(den2, 0, (size_t)N_NODES * sizeof(float), stream);
  hipMemsetAsync(out2, 0, (size_t)N_NODES * OUT_C * sizeof(float), stream);

  gemm2_k<<<N_NODES / 16, 128, 0, stream>>>(out1, w2p, h2);
  attn2_k<<<N_NODES / 8, 256, 0, stream>>>(h2, att_s2, att_d2, as2, ad2);
  ep1_l2<<<(ETOT + 255) / 256, 256, 0, stream>>>(ei, as2, ad2, m2u);
  ep2_l2<<<(ETOT + 255) / 256, 256, 0, stream>>>(ei, as2, ad2, m2u, den2);
  ep3_l2<<<ETOT / 8, 256, 0, stream>>>(ei, as2, ad2, m2u, den2, h2, out2);
  final_k<<<N_NODES / 8, 256, 0, stream>>>(out2, b2, y);
}
